// Encoder_41412074668419
// MI455X (gfx1250) — matrix operand order repaired and checked by reading, not compiled, not run on hardware
//
#include <hip/hip_runtime.h>
#include <hip/hip_bf16.h>
#include <math.h>

// ---- problem constants ------------------------------------------------------
#define BB      8          // batch
#define LL      256        // time length
#define EE      41         // channels
#define NKDIM   128        // model dim
#define NHEADS  4
#define NOBS    2624       // observations per row
#define LEFULL  (LL*EE)    // 10496 = 41*256
#define ECHPAD  48         // channel rows padded to multiple of 16

typedef __bf16 bf16;
typedef bf16  v16bf __attribute__((ext_vector_type(16)));
typedef bf16  v8bf  __attribute__((ext_vector_type(8)));
typedef float v8f   __attribute__((ext_vector_type(8)));

__device__ __forceinline__ v8f wmma_bf16(v16bf a, v16bf b, v8f c) {
  // D = A(16x32 bf16) x B(32x16 bf16) + C(16x16 f32)
  return __builtin_amdgcn_wmma_f32_16x16x32_bf16(false, a, false, b, (short)0, c,
                                                 false, false);
}

// ---- 1) ragged scan: observed positions -> idx/tif/cif + tv_f/tm_f outputs --
__global__ __launch_bounds__(256)
void scan_kernel(const float* __restrict__ mask, const float* __restrict__ tval,
                 const float* __restrict__ tmask,
                 int* __restrict__ idxp, int* __restrict__ tif, int* __restrict__ cif,
                 float* __restrict__ out_tv, float* __restrict__ out_tm)
{
  __shared__ int sdat[256];
  const int b = blockIdx.x;
  const int t = threadIdx.x;
  int base = 0;
  for (int c = 0; c < LEFULL / 256; ++c) {
    int p = c * 256 + t;
    int f = (mask[(size_t)b * LEFULL + p] > 0.5f) ? 1 : 0;
    sdat[t] = f;
    __syncthreads();
    for (int off = 1; off < 256; off <<= 1) {
      int v = (t >= off) ? sdat[t - off] : 0;
      __syncthreads();
      sdat[t] += v;
      __syncthreads();
    }
    int rank = base + sdat[t] - 1;
    if (f && rank < NOBS) {
      size_t o = (size_t)b * NOBS + rank;
      idxp[o] = p;
      tif[o]  = p / EE;
      cif[o]  = p % EE;
      out_tv[o] = tval[(size_t)b * LEFULL + p];
      out_tm[o] = tmask[(size_t)b * LEFULL + p];
    }
    base += sdat[255];
    __syncthreads();
  }
}

// ---- 2) weight convert f32 [K,128] -> bf16 transposed [128,K] --------------
__global__ __launch_bounds__(256)
void wconv_kernel(const float* __restrict__ W, bf16* __restrict__ Wt, int K)
{
  int g = blockIdx.x * 256 + threadIdx.x;
  if (g < K * NKDIM) {
    int k = g / NKDIM, c = g % NKDIM;
    Wt[(size_t)c * K + k] = (bf16)W[g];
  }
}

// ---- 3) init: value_flat / time_emb / chan_emb (f32 + bf16) -----------------
__global__ __launch_bounds__(256)
void init_kernel(const float* __restrict__ value, const float* __restrict__ tmask,
                 const float* __restrict__ cx, const int* __restrict__ idxp,
                 const float* __restrict__ We, const float* __restrict__ be,
                 const float* __restrict__ Wtm, const float* __restrict__ btm,
                 const float* __restrict__ Wc, const float* __restrict__ bc,
                 float* __restrict__ vf_f, bf16* __restrict__ vf_h,
                 float* __restrict__ te_f, bf16* __restrict__ te_h,
                 float* __restrict__ ce_f, bf16* __restrict__ ce_h)
{
  const size_t vtot = (size_t)BB * NOBS * NKDIM;
  const size_t ttot = (size_t)BB * LL * NKDIM;
  const size_t ctot = (size_t)BB * ECHPAD * NKDIM;
  size_t g = (size_t)blockIdx.x * 256 + threadIdx.x;
  if (g < vtot) {
    int k = (int)(g & 127); size_t bn = g >> 7;
    int b = (int)(bn / NOBS);
    int p = idxp[bn];
    float v  = value[(size_t)b * LEFULL + p];
    float tm = tmask[(size_t)b * LEFULL + p];     // lookback == tm (mask_f == 1)
    float r = fmaxf(v * We[k] + tm * We[NKDIM + k] + be[k], 0.f);
    vf_f[g] = r; vf_h[g] = (bf16)r;
  } else if (g < vtot + ttot) {
    size_t gg = g - vtot;
    int k = (int)(gg & 127); size_t bl = gg >> 7;
    float r = sinf(cx[bl] * Wtm[k] + btm[k]);
    te_f[gg] = r; te_h[gg] = (bf16)r;
  } else if (g < vtot + ttot + ctot) {
    size_t gg = g - vtot - ttot;
    int k = (int)(gg & 127); size_t be_i = gg >> 7; int e = (int)(be_i % ECHPAD);
    float r = (e < EE) ? fmaxf(Wc[(size_t)e * NKDIM + k] + bc[k], 0.f) : 0.f;
    ce_f[gg] = r; ce_h[gg] = (bf16)r;
  }
}

// ---- 4) generic WMMA GEMM ---------------------------------------------------
// One width-128 bf16 segment, optional per-batch row-gather, accumulated via
// 4 WMMA K-steps. Called with distinct kernel pointer args so address-space
// inference yields global_load_b128 (not flat_load: FLAT burns DScnt + LDS path).
__device__ __forceinline__ void gemm_seg(v8f& acc, const bf16* __restrict__ A,
                                         const int* __restrict__ ix, int rows,
                                         const bf16* __restrict__ wseg,
                                         int b, int mrow, int hi)
{
  int arow = ix ? ix[(size_t)b * NOBS + mrow] : mrow;
  const bf16* abase = A + ((size_t)b * rows + arow) * NKDIM + hi * 16;
#pragma unroll
  for (int kb = 0; kb < 4; ++kb) {
    v8bf alo = *(const v8bf*)(abase + kb * 32);
    v8bf ahi = *(const v8bf*)(abase + kb * 32 + 8);
    v8bf blo = *(const v8bf*)(wseg + kb * 32);
    v8bf bhi = *(const v8bf*)(wseg + kb * 32 + 8);
    v16bf af, bfv;
#pragma unroll
    for (int e = 0; e < 8; ++e) {
      af[e] = alo[e];  af[e + 8]  = ahi[e];
      bfv[e] = blo[e]; bfv[e + 8] = bhi[e];
    }
    acc = wmma_bf16(af, bfv, acc);
  }
}

// C[M,128] = ep(res, concat(A0..A2) @ W + bias)
//  * Wt is bf16, transposed: Wt[c*Ktot + k]
//  * ep: 0 -> x ; 1 -> res + relu(x) ; 2 -> relu(res + x)
//  * Cf (f32), Ch (bf16 row-major), ChT (bf16 transposed [128 x M]) all nullable
__global__ __launch_bounds__(256)
void gemm_kernel(const bf16* __restrict__ A0, const int* __restrict__ i0, int rows0,
                 const bf16* __restrict__ A1, const int* __restrict__ i1, int rows1,
                 const bf16* __restrict__ A2, const int* __restrict__ i2, int rows2,
                 int nseg, const bf16* __restrict__ Wt, int Ktot,
                 const float* __restrict__ bias,
                 const float* __restrict__ res, int ep,
                 float* __restrict__ Cf, bf16* __restrict__ Ch,
                 bf16* __restrict__ ChT, int M)
{
  const int b    = blockIdx.y;
  const int mt   = blockIdx.x;
  const int wave = threadIdx.x >> 5;
  const int lane = threadIdx.x & 31;
  const int hi   = lane >> 4;
  const int nl   = lane & 15;
  const int ccol = wave * 16 + nl;
  const int mrow = mt * 16 + nl;          // A-fragment row for this lane

  v8f acc = {0.f,0.f,0.f,0.f,0.f,0.f,0.f,0.f};
  const bf16* wb = Wt + (size_t)ccol * Ktot + hi * 16;

  gemm_seg(acc, A0, i0, rows0, wb, b, mrow, hi);
  if (nseg > 1) gemm_seg(acc, A1, i1, rows1, wb + NKDIM, b, mrow, hi);
  if (nseg > 2) gemm_seg(acc, A2, i2, rows2, wb + 2 * NKDIM, b, mrow, hi);

  __syncthreads();                        // in-place safety for edge update
  float bv = bias[ccol];
#pragma unroll
  for (int r = 0; r < 8; ++r) {
    int m = mt * 16 + r + hi * 8;
    size_t off = ((size_t)b * M + m) * NKDIM + ccol;
    float v = acc[r] + bv;
    if (ep == 1)      v = res[off] + fmaxf(v, 0.f);
    else if (ep == 2) v = fmaxf(res[off] + v, 0.f);
    if (Cf)  Cf[off] = v;
    if (Ch)  Ch[off] = (bf16)v;
    if (ChT) ChT[((size_t)b * NKDIM + ccol) * M + m] = (bf16)v;
  }
}

// ---- 5) masked flash attention, 4 heads x d=32, WMMA score & PV -------------
// VhT is the transposed V: [B][128(d)][NOBS(keys)] so PV B-fragments are
// contiguous 16B loads along the key (K) dimension.
__global__ __launch_bounds__(128)
void attn_kernel(const bf16* __restrict__ Qh, const float* __restrict__ Qf,
                 const bf16* __restrict__ Kh, const bf16* __restrict__ VhT,
                 const int* __restrict__ keysel, int Mq,
                 float* __restrict__ Of, bf16* __restrict__ Oh)
{
  __shared__ float Plds[NHEADS][16][32];
  const int b  = blockIdx.y;
  const int qt = blockIdx.x;
  const int h    = threadIdx.x >> 5;
  const int lane = threadIdx.x & 31;
  const int hi   = lane >> 4;
  const int nl   = lane & 15;
  const float scale = 0.08838834764831845f;   // 1/sqrt(NK=128)

  // Q fragment (A layout), rows qt*16+nl, K-dim = head slice d = h*32..+31
  v16bf qfrag;
  {
    const bf16* qb = Qh + ((size_t)b * Mq + qt * 16 + nl) * NKDIM + h * 32 + hi * 16;
    v8bf qlo = *(const v8bf*)(qb);
    v8bf qhi = *(const v8bf*)(qb + 8);
#pragma unroll
    for (int e = 0; e < 8; ++e) { qfrag[e] = qlo[e]; qfrag[e + 8] = qhi[e]; }
  }

  v8f acc0 = {0.f,0.f,0.f,0.f,0.f,0.f,0.f,0.f};
  v8f acc1 = {0.f,0.f,0.f,0.f,0.f,0.f,0.f,0.f};
  float runmax[8], runsum[8];
#pragma unroll
  for (int r = 0; r < 8; ++r) { runmax[r] = -1e30f; runsum[r] = 0.f; }

  const int* ks = keysel + (size_t)b * NOBS;
  // transposed-V row bases for the two 16-wide d tiles of this head
  const bf16* vt0 = VhT + ((size_t)b * NKDIM + h * 32 + nl)      * NOBS + hi * 16;
  const bf16* vt1 = VhT + ((size_t)b * NKDIM + h * 32 + 16 + nl) * NOBS + hi * 16;

  for (int kc = 0; kc < NOBS; kc += 32) {
    // K^T fragments (B layout): cols = keys, K-dim = d
    v16bf k0, k1;
    {
      const bf16* kb0 = Kh + ((size_t)b * NOBS + kc + nl) * NKDIM + h * 32 + hi * 16;
      const bf16* kb1 = kb0 + (size_t)16 * NKDIM;
      __builtin_prefetch(kb0 + (size_t)32 * NKDIM, 0, 0);
      v8bf a0 = *(const v8bf*)(kb0), a1 = *(const v8bf*)(kb0 + 8);
      v8bf c0 = *(const v8bf*)(kb1), c1 = *(const v8bf*)(kb1 + 8);
#pragma unroll
      for (int e = 0; e < 8; ++e) {
        k0[e] = a0[e]; k0[e + 8] = a1[e];
        k1[e] = c0[e]; k1[e + 8] = c1[e];
      }
    }
    v8f z = {0.f,0.f,0.f,0.f,0.f,0.f,0.f,0.f};
    v8f s0 = wmma_bf16(qfrag, k0, z);
    v8f s1 = wmma_bf16(qfrag, k1, z);

    int sel0 = ks[kc + nl];
    int sel1 = ks[kc + 16 + nl];
#pragma unroll
    for (int r = 0; r < 8; ++r) {
      int qrow = qt * 16 + r + hi * 8;
      float a = (sel0 == qrow) ? s0[r] * scale : -1e10f;   // exact ref mask value
      float c = (sel1 == qrow) ? s1[r] * scale : -1e10f;
      float cm = fmaxf(a, c);
      cm = fmaxf(cm, __shfl_xor(cm, 1, 32));
      cm = fmaxf(cm, __shfl_xor(cm, 2, 32));
      cm = fmaxf(cm, __shfl_xor(cm, 4, 32));
      cm = fmaxf(cm, __shfl_xor(cm, 8, 32));
      float nmax = fmaxf(runmax[r], cm);
      float sc = __expf(runmax[r] - nmax);
      float e0 = __expf(a - nmax);
      float e1 = __expf(c - nmax);
      float cs = e0 + e1;
      cs += __shfl_xor(cs, 1, 32);
      cs += __shfl_xor(cs, 2, 32);
      cs += __shfl_xor(cs, 4, 32);
      cs += __shfl_xor(cs, 8, 32);
      runsum[r] = runsum[r] * sc + cs;
      runmax[r] = nmax;
      acc0[r] *= sc; acc1[r] *= sc;
      Plds[h][r + hi * 8][nl]      = e0;
      Plds[h][r + hi * 8][nl + 16] = e1;
    }
    __syncthreads();
    // P fragment (A layout): rows nl, K-dim over 32 keys
    v16bf pf;
#pragma unroll
    for (int e = 0; e < 16; ++e) {
      int kk = e + hi * 16;
      pf[e] = (bf16)Plds[h][nl][kk];
    }
    // V fragments (B layout) from transposed V: contiguous along keys
    v16bf v0, v1;
    {
      v8bf a0 = *(const v8bf*)(vt0 + kc), a1 = *(const v8bf*)(vt0 + kc + 8);
      v8bf c0 = *(const v8bf*)(vt1 + kc), c1 = *(const v8bf*)(vt1 + kc + 8);
#pragma unroll
      for (int e = 0; e < 8; ++e) {
        v0[e] = a0[e]; v0[e + 8] = a1[e];
        v1[e] = c0[e]; v1[e + 8] = c1[e];
      }
    }
    acc0 = wmma_bf16(pf, v0, acc0);
    acc1 = wmma_bf16(pf, v1, acc1);
    __syncthreads();
  }

#pragma unroll
  for (int r = 0; r < 8; ++r) {
    int m = qt * 16 + r + hi * 8;
    float inv = 1.f / runsum[r];
    size_t o0 = ((size_t)b * Mq + m) * NKDIM + h * 32 + nl;
    float a0 = acc0[r] * inv + Qf[o0];        // o = qh + A@vh
    float a1 = acc1[r] * inv + Qf[o0 + 16];
    Of[o0] = a0;      Oh[o0] = (bf16)a0;
    Of[o0 + 16] = a1; Oh[o0 + 16] = (bf16)a1;
  }
}

// ---- 6) final output: dot(concat(vf, k_t, k_c), Wout) + bout ---------------
__global__ __launch_bounds__(256)
void out_kernel(const float* __restrict__ vf, const float* __restrict__ te,
                const float* __restrict__ ce, const int* __restrict__ tif,
                const int* __restrict__ cif, const float* __restrict__ Wout,
                const float* __restrict__ bout, float* __restrict__ out)
{
  int g = blockIdx.x * 256 + threadIdx.x;
  if (g >= BB * NOBS) return;
  int b = g / NOBS;
  int t = tif[g], c = cif[g];
  const float* vrow = vf + (size_t)g * NKDIM;
  const float* trow = te + ((size_t)b * LL + t) * NKDIM;
  const float* crow = ce + ((size_t)b * ECHPAD + c) * NKDIM;
  float acc = bout[0];
#pragma unroll 4
  for (int k = 0; k < NKDIM; ++k)
    acc += vrow[k] * Wout[k] + trow[k] * Wout[NKDIM + k] + crow[k] * Wout[2 * NKDIM + k];
  out[g] = acc;
}

// ---- host side --------------------------------------------------------------
extern "C" void kernel_launch(void* const* d_in, const int* in_sizes, int n_in,
                              void* d_out, int out_size, void* d_ws, size_t ws_size,
                              hipStream_t stream)
{
  (void)in_sizes; (void)n_in; (void)out_size; (void)ws_size;
  const float* cx    = (const float*)d_in[0];
  const float* value = (const float*)d_in[1];
  const float* mask  = (const float*)d_in[2];
  const float* tval  = (const float*)d_in[3];
  const float* tmask = (const float*)d_in[4];
  // d_in[5] = exp_stage (unused by reference)
  auto F = [&](int i) { return (const float*)d_in[i]; };
  // pytree (alphabetical) param order:
  // 6:chan_W 7:chan_b 8:edge_W 9:edge_b
  // per layer (base LB=10+18*l): ct.k.W/b, ct.o.W/b, ct.q.W/b, ct.v.W/b,
  //                              edge_nn.W/b, tc.k.W/b, tc.o.W/b, tc.q.W/b, tc.v.W/b
  // 64:out_W 65:out_b 66:time_W 67:time_b

  // workspace carve (256B aligned)
  char* wbase = (char*)d_ws;
  size_t woff = 0;
  auto carve = [&](size_t bytes) -> char* {
    char* p = wbase + woff;
    woff = (woff + bytes + 255) & ~(size_t)255;
    return p;
  };
  int* idxp = (int*)carve((size_t)BB * NOBS * 4);
  int* tif  = (int*)carve((size_t)BB * NOBS * 4);
  int* cif  = (int*)carve((size_t)BB * NOBS * 4);
  float* vf_f = (float*)carve((size_t)BB * NOBS * NKDIM * 4);
  bf16*  vf_h = (bf16*) carve((size_t)BB * NOBS * NKDIM * 2);
  float* te_f = (float*)carve((size_t)BB * LL * NKDIM * 4);
  bf16*  te_h = (bf16*) carve((size_t)BB * LL * NKDIM * 2);
  float* ce_f = (float*)carve((size_t)BB * ECHPAD * NKDIM * 4);
  bf16*  ce_h = (bf16*) carve((size_t)BB * ECHPAD * NKDIM * 2);
  float* qc_f = (float*)carve((size_t)BB * ECHPAD * NKDIM * 4);
  bf16*  qc_h = (bf16*) carve((size_t)BB * ECHPAD * NKDIM * 2);
  float* qt_f = (float*)carve((size_t)BB * LL * NKDIM * 4);
  bf16*  qt_h = (bf16*) carve((size_t)BB * LL * NKDIM * 2);
  bf16*  kc_h = (bf16*) carve((size_t)BB * NOBS * NKDIM * 2);
  bf16*  vcT  = (bf16*) carve((size_t)BB * NOBS * NKDIM * 2);  // transposed V
  bf16*  kt_h = (bf16*) carve((size_t)BB * NOBS * NKDIM * 2);
  bf16*  vtT  = (bf16*) carve((size_t)BB * NOBS * NKDIM * 2);  // transposed V
  float* Oc_f = (float*)carve((size_t)BB * ECHPAD * NKDIM * 4);
  bf16*  Oc_h = (bf16*) carve((size_t)BB * ECHPAD * NKDIM * 2);
  float* Ot_f = (float*)carve((size_t)BB * LL * NKDIM * 4);
  bf16*  Ot_h = (bf16*) carve((size_t)BB * LL * NKDIM * 2);
  // transposed bf16 weights: per layer {ctq, ctk, ctv, cto, edge, tcq, tck, tcv, tco}
  bf16* wt[3][9];
  const int wk[9] = {128, 256, 256, 128, 384, 128, 256, 256, 128};
  for (int l = 0; l < 3; ++l)
    for (int j = 0; j < 9; ++j)
      wt[l][j] = (bf16*)carve((size_t)wk[j] * NKDIM * 2);

  float* out0   = (float*)d_out;
  float* out_tv = out0 + (size_t)BB * NOBS;
  float* out_tm = out0 + 2 * (size_t)BB * NOBS;

  // 1) scan -> idx/tif/cif + tv_f/tm_f outputs
  scan_kernel<<<BB, 256, 0, stream>>>(mask, tval, tmask, idxp, tif, cif, out_tv, out_tm);

  // 2) weight conversion (transposed bf16)
  for (int l = 0; l < 3; ++l) {
    int LB = 10 + 18 * l;
    const int widx[9] = {LB + 4, LB + 0, LB + 6, LB + 2, LB + 8,
                         LB + 14, LB + 10, LB + 16, LB + 12};
    for (int j = 0; j < 9; ++j) {
      int n = wk[j] * NKDIM;
      wconv_kernel<<<(n + 255) / 256, 256, 0, stream>>>(F(widx[j]), wt[l][j], wk[j]);
    }
  }

  // 3) embeddings init
  {
    size_t tot = (size_t)BB * NOBS * NKDIM + (size_t)BB * LL * NKDIM
               + (size_t)BB * ECHPAD * NKDIM;
    init_kernel<<<(unsigned)((tot + 255) / 256), 256, 0, stream>>>(
        value, tmask, cx, idxp, F(8), F(9), F(66), F(67), F(6), F(7),
        vf_f, vf_h, te_f, te_h, ce_f, ce_h);
  }

  const dim3 gObs(NOBS / 16, BB);   // 164 x 8
  const dim3 gCh(ECHPAD / 16, BB);  // 3 x 8
  const dim3 gTm(LL / 16, BB);      // 16 x 8

  for (int l = 0; l < 3; ++l) {
    int LB = 10 + 18 * l;
    // q projections
    gemm_kernel<<<gCh, 256, 0, stream>>>(ce_h, nullptr, ECHPAD, nullptr, nullptr, 0,
        nullptr, nullptr, 0, 1, wt[l][0], 128, F(LB + 5), nullptr, 0,
        qc_f, qc_h, nullptr, ECHPAD);
    gemm_kernel<<<gTm, 256, 0, stream>>>(te_h, nullptr, LL, nullptr, nullptr, 0,
        nullptr, nullptr, 0, 1, wt[l][5], 128, F(LB + 15), nullptr, 0,
        qt_f, qt_h, nullptr, LL);
    // k/v projections on concat([k_t, value_flat]) and concat([k_c, value_flat])
    gemm_kernel<<<gObs, 256, 0, stream>>>(te_h, tif, LL, vf_h, nullptr, NOBS,
        nullptr, nullptr, 0, 2, wt[l][1], 256, F(LB + 1), nullptr, 0,
        nullptr, kc_h, nullptr, NOBS);
    gemm_kernel<<<gObs, 256, 0, stream>>>(te_h, tif, LL, vf_h, nullptr, NOBS,
        nullptr, nullptr, 0, 2, wt[l][2], 256, F(LB + 7), nullptr, 0,
        nullptr, nullptr, vcT, NOBS);
    gemm_kernel<<<gObs, 256, 0, stream>>>(ce_h, cif, ECHPAD, vf_h, nullptr, NOBS,
        nullptr, nullptr, 0, 2, wt[l][6], 256, F(LB + 11), nullptr, 0,
        nullptr, kt_h, nullptr, NOBS);
    gemm_kernel<<<gObs, 256, 0, stream>>>(ce_h, cif, ECHPAD, vf_h, nullptr, NOBS,
        nullptr, nullptr, 0, 2, wt[l][7], 256, F(LB + 17), nullptr, 0,
        nullptr, nullptr, vtT, NOBS);
    // masked attention (CT: channel queries vs obs keys; TC: time queries)
    attn_kernel<<<gCh, 128, 0, stream>>>(qc_h, qc_f, kc_h, vcT, cif, ECHPAD, Oc_f, Oc_h);
    attn_kernel<<<gTm, 128, 0, stream>>>(qt_h, qt_f, kt_h, vtT, tif, LL, Ot_f, Ot_h);
    // edge update: vf = relu(vf + concat(vf, k_t, k_c) @ We + be)   (uses OLD embs)
    gemm_kernel<<<gObs, 256, 0, stream>>>(vf_h, nullptr, NOBS, te_h, tif, LL,
        ce_h, cif, ECHPAD, 3, wt[l][4], 384, F(LB + 9), vf_f, 2,
        vf_f, vf_h, nullptr, NOBS);
    // MAB outputs: emb = o + relu(o @ Wo + bo)  -> become next-layer embeddings
    gemm_kernel<<<gCh, 256, 0, stream>>>(Oc_h, nullptr, ECHPAD, nullptr, nullptr, 0,
        nullptr, nullptr, 0, 1, wt[l][3], 128, F(LB + 3), Oc_f, 1,
        ce_f, ce_h, nullptr, ECHPAD);
    gemm_kernel<<<gTm, 256, 0, stream>>>(Ot_h, nullptr, LL, nullptr, nullptr, 0,
        nullptr, nullptr, 0, 1, wt[l][8], 128, F(LB + 13), Ot_f, 1,
        te_f, te_h, nullptr, LL);
  }

  // final: out = concat(vf, k_t, k_c) @ Wout + bout
  out_kernel<<<(BB * NOBS + 255) / 256, 256, 0, stream>>>(
      vf_f, te_f, ce_f, tif, cif, F(64), F(65), out0);
}
